// TransformerBlock_14671608283841
// MI455X (gfx1250) — matrix operand order repaired and checked by reading, not compiled, not run on hardware
//
#include <hip/hip_runtime.h>

typedef __attribute__((ext_vector_type(16))) _Float16 v16h;
typedef __attribute__((ext_vector_type(8)))  _Float16 v8h;
typedef __attribute__((ext_vector_type(8)))  float    v8f;

#define WMMA16(a, b, c) __builtin_amdgcn_wmma_f32_16x16x32_f16( \
    false, (a), false, (b), (short)0, (c), false, false)

static __device__ __forceinline__ v16h load_a_frag(const _Float16* p, int ld) {
  // A-matrix 16x32 f16: lane&15 = row M; chunks K=[koff, koff+8) and K=[koff+16, koff+24)
  int lane = threadIdx.x & 31;
  int m    = lane & 15;
  int koff = (lane >> 4) << 3;
  const _Float16* r = p + (size_t)m * ld + koff;
  union { v16h v; v8h h[2]; } u;
  u.h[0] = *(const v8h*)(r);
  u.h[1] = *(const v8h*)(r + 16);
  return u.v;
}

static __device__ __forceinline__ v16h load_b_frag(const _Float16* p, int ld) {
  // B-matrix 32x16 f16: lane&15 = col N; chunks K=[koff, koff+8) and K=[koff+16, koff+24)
  int lane = threadIdx.x & 31;
  int n    = lane & 15;
  int koff = (lane >> 4) << 3;
  const _Float16* r = p + (size_t)n * ld + koff;
  union { v16h v; v8h h[2]; } u;
  u.h[0] = *(const v8h*)(r);
  u.h[1] = *(const v8h*)(r + 16);
  return u.v;
}

// ---------------------------------------------------------------- utilities
__global__ void cvt_f32_f16(const float* __restrict__ in, _Float16* __restrict__ out, int n) {
  int i = blockIdx.x * blockDim.x + threadIdx.x;
  int stride = gridDim.x * blockDim.x;
  for (; i < n; i += stride) out[i] = (_Float16)in[i];
}

__global__ __launch_bounds__(256) void rmsnorm_f16(const float* __restrict__ x,
                                                   const float* __restrict__ w,
                                                   _Float16* __restrict__ out, int dim) {
  int row = blockIdx.x;
  const float* xr = x + (size_t)row * dim;
  float ss = 0.f;
  for (int i = threadIdx.x; i < dim; i += 256) { float v = xr[i]; ss += v * v; }
  for (int off = 16; off >= 1; off >>= 1) ss += __shfl_xor(ss, off, 32);
  __shared__ float red[8];
  int wave = threadIdx.x >> 5, lane = threadIdx.x & 31;
  if (lane == 0) red[wave] = ss;
  __syncthreads();
  if (threadIdx.x == 0) {
    float t = 0.f;
    for (int i = 0; i < 8; ++i) t += red[i];
    red[0] = rsqrtf(t / (float)dim + 1e-5f);
  }
  __syncthreads();
  float scale = red[0];
  for (int i = threadIdx.x; i < dim; i += 256)
    out[(size_t)row * dim + i] = (_Float16)(xr[i] * scale * w[i]);
}

__global__ void silu_mul_f16(const _Float16* __restrict__ a, const _Float16* __restrict__ b,
                             _Float16* __restrict__ out, int n) {
  int i = blockIdx.x * blockDim.x + threadIdx.x;
  int stride = gridDim.x * blockDim.x;
  for (; i < n; i += stride) {
    float av = (float)a[i], bv = (float)b[i];
    out[i] = (_Float16)((av / (1.f + __expf(-av))) * bv);
  }
}

// ------------------------------------------------------------- WMMA GEMM
// C[M,N] = A[M,K](f16,row-major) * W[N,K](f16,row-major)^T ; f32 accumulate.
// Block: 256 threads = 8 waves (2x4); block tile 128x256; wave tile 64x64.
// Per k-step: 8 fragment loads feed 16 WMMAs (16 B/lane/WMMA cache traffic).
__global__ __launch_bounds__(256) void gemm_wmma(const _Float16* __restrict__ A,
                                                 const _Float16* __restrict__ W,
                                                 const float* __restrict__ R,
                                                 float* __restrict__ Cf,
                                                 _Float16* __restrict__ Ch,
                                                 int N, int K) {
  const int wave = threadIdx.x >> 5;
  const int lane = threadIdx.x & 31;
  const int wm = wave >> 2;            // 0..1
  const int wn = wave & 3;             // 0..3
  const int row0 = blockIdx.y * 128 + wm * 64;
  const int col0 = blockIdx.x * 256 + wn * 64;

  const v8f vzero = {0.f, 0.f, 0.f, 0.f, 0.f, 0.f, 0.f, 0.f};
  v8f acc[4][4];
#pragma unroll
  for (int i = 0; i < 4; ++i)
#pragma unroll
    for (int j = 0; j < 4; ++j) acc[i][j] = vzero;

  for (int k = 0; k < K; k += 32) {
    v16h a[4];
#pragma unroll
    for (int mt = 0; mt < 4; ++mt)
      a[mt] = load_a_frag(A + (size_t)(row0 + mt * 16) * K + k, K);
#pragma unroll
    for (int nt = 0; nt < 4; ++nt) {
      v16h b = load_b_frag(W + (size_t)(col0 + nt * 16) * K + k, K);
#pragma unroll
      for (int mt = 0; mt < 4; ++mt)
        acc[mt][nt] = WMMA16(a[mt], b, acc[mt][nt]);
    }
  }

  const int n  = lane & 15;
  const int mb = (lane >> 4) << 3;
  if (Cf) {
#pragma unroll
    for (int mt = 0; mt < 4; ++mt)
#pragma unroll
      for (int nt = 0; nt < 4; ++nt)
#pragma unroll
        for (int r = 0; r < 8; ++r) {
          size_t idx = (size_t)(row0 + mt * 16 + mb + r) * N + (col0 + nt * 16 + n);
          float v = acc[mt][nt][r];
          if (R) v += R[idx];
          Cf[idx] = v;
        }
  } else {
#pragma unroll
    for (int mt = 0; mt < 4; ++mt)
#pragma unroll
      for (int nt = 0; nt < 4; ++nt)
#pragma unroll
        for (int r = 0; r < 8; ++r) {
          size_t idx = (size_t)(row0 + mt * 16 + mb + r) * N + (col0 + nt * 16 + n);
          Ch[idx] = (_Float16)acc[mt][nt][r];
        }
  }
}

// ------------------------------------------------------------------ RoPE
// qkv: (B*S, 6144) f16 rows. Outputs q,k in (B,H,S,128); v transposed (B,H,128,S).
// Reference rope: out[2i] = x[i]*cos - x[64+i]*sin ; out[2i+1] = x[64+i]*cos + x[i]*sin
__global__ __launch_bounds__(128) void rope_split(const _Float16* __restrict__ qkv,
                                                  const float* __restrict__ freqs,
                                                  _Float16* __restrict__ qh,
                                                  _Float16* __restrict__ kh,
                                                  _Float16* __restrict__ vt) {
  int s = blockIdx.x, h = blockIdx.y, b = blockIdx.z;
  size_t rowoff = ((size_t)b * 2048 + s) * 6144;
  const _Float16* qrow = qkv + rowoff + h * 128;
  const _Float16* krow = qkv + rowoff + 2048 + h * 128;
  const _Float16* vrow = qkv + rowoff + 4096 + h * 128;
  size_t bh = (size_t)(b * 16 + h);
  _Float16* qo = qh + (bh * 2048 + s) * 128;
  _Float16* ko = kh + (bh * 2048 + s) * 128;
  int t = threadIdx.x;
  if (t < 64) {
    float cs = freqs[((size_t)s * 64 + t) * 2 + 0];
    float sn = freqs[((size_t)s * 64 + t) * 2 + 1];
    float q1 = (float)qrow[t], q2 = (float)qrow[64 + t];
    qo[2 * t]     = (_Float16)(q1 * cs - q2 * sn);
    qo[2 * t + 1] = (_Float16)(q2 * cs + q1 * sn);
    float k1 = (float)krow[t], k2 = (float)krow[64 + t];
    ko[2 * t]     = (_Float16)(k1 * cs - k2 * sn);
    ko[2 * t + 1] = (_Float16)(k2 * cs + k1 * sn);
  }
  vt[(bh * 128 + t) * 2048 + s] = vrow[t];   // transpose v: (d, s)
}

// --------------------------------------------------------- Flash attention
// grid (S/64, H, B), block 128 = 4 waves; each wave owns 16 q-rows.
__global__ __launch_bounds__(128) void flash_attn(const _Float16* __restrict__ qh,
                                                  const _Float16* __restrict__ kh,
                                                  const _Float16* __restrict__ vt,
                                                  _Float16* __restrict__ out) {
  const float SCALE = 0.08838834764831845f;   // 1/sqrt(128)
  const float NEG_INF = -3.4028234663852886e38f;
  const int b = blockIdx.z, h = blockIdx.y;
  const int wave = threadIdx.x >> 5, lane = threadIdx.x & 31;
  const int q0 = (blockIdx.x * 4 + wave) * 16;
  const size_t bh = (size_t)(b * 16 + h);
  const _Float16* qp = qh + (bh * 2048 + q0) * 128;
  const _Float16* kp = kh + bh * 2048 * 128;
  const _Float16* vp = vt + bh * 128 * 2048;

  __shared__ _Float16 Plds[4][16][32];

  const v8f vzero = {0.f, 0.f, 0.f, 0.f, 0.f, 0.f, 0.f, 0.f};
  v16h qf[4];
#pragma unroll
  for (int kk = 0; kk < 4; ++kk) qf[kk] = load_a_frag(qp + kk * 32, 128);

  v8f o[8];
#pragma unroll
  for (int t = 0; t < 8; ++t) o[t] = vzero;
  float mrun[8], lrun[8];
#pragma unroll
  for (int r = 0; r < 8; ++r) { mrun[r] = NEG_INF; lrun[r] = 0.f; }

  const int n  = lane & 15;
  const int mb = (lane >> 4) << 3;

  for (int j = 0; j <= q0 + 15; j += 32) {
    v8f s0 = vzero, s1 = vzero;
#pragma unroll
    for (int kk = 0; kk < 4; ++kk) {
      v16h b0 = load_b_frag(kp + (size_t)j * 128 + kk * 32, 128);
      v16h b1 = load_b_frag(kp + (size_t)(j + 16) * 128 + kk * 32, 128);
      s0 = WMMA16(qf[kk], b0, s0);
      s1 = WMMA16(qf[kk], b1, s1);
    }
    float p0[8], p1[8], rowm[8];
#pragma unroll
    for (int r = 0; r < 8; ++r) {
      int row = q0 + mb + r;
      float a = s0[r] * SCALE; if (j + n      > row) a = NEG_INF;
      float c = s1[r] * SCALE; if (j + 16 + n > row) c = NEG_INF;
      p0[r] = a; p1[r] = c;
      rowm[r] = fmaxf(a, c);
    }
#pragma unroll
    for (int off = 1; off < 16; off <<= 1)
#pragma unroll
      for (int r = 0; r < 8; ++r) rowm[r] = fmaxf(rowm[r], __shfl_xor(rowm[r], off, 16));

    float alpha[8], rs[8];
#pragma unroll
    for (int r = 0; r < 8; ++r) {
      float mnew = fmaxf(mrun[r], rowm[r]);
      alpha[r] = __expf(mrun[r] - mnew);
      mrun[r] = mnew;
      p0[r] = __expf(p0[r] - mnew);
      p1[r] = __expf(p1[r] - mnew);
      rs[r] = p0[r] + p1[r];
    }
#pragma unroll
    for (int off = 1; off < 16; off <<= 1)
#pragma unroll
      for (int r = 0; r < 8; ++r) rs[r] += __shfl_xor(rs[r], off, 16);
#pragma unroll
    for (int r = 0; r < 8; ++r) lrun[r] = lrun[r] * alpha[r] + rs[r];
#pragma unroll
    for (int t = 0; t < 8; ++t)
#pragma unroll
      for (int r = 0; r < 8; ++r) o[t][r] *= alpha[r];

    // stage P (D-layout -> A-layout) through per-wave LDS
#pragma unroll
    for (int r = 0; r < 8; ++r) {
      Plds[wave][mb + r][n]      = (_Float16)p0[r];
      Plds[wave][mb + r][16 + n] = (_Float16)p1[r];
    }
    asm volatile("s_wait_dscnt 0" ::: "memory");
    v16h pf = load_a_frag(&Plds[wave][0][0], 32);
#pragma unroll
    for (int dt = 0; dt < 8; ++dt) {
      v16h vb = load_b_frag(vp + (size_t)(dt * 16) * 2048 + j, 2048);
      o[dt] = WMMA16(pf, vb, o[dt]);
    }
  }

  float inv[8];
#pragma unroll
  for (int r = 0; r < 8; ++r) inv[r] = 1.f / lrun[r];
#pragma unroll
  for (int dt = 0; dt < 8; ++dt)
#pragma unroll
    for (int r = 0; r < 8; ++r) {
      int row = q0 + mb + r;
      out[((size_t)b * 2048 + row) * 2048 + h * 128 + dt * 16 + n] =
          (_Float16)(o[dt][r] * inv[r]);
    }
}

// ------------------------------------------------------------------ driver
extern "C" void kernel_launch(void* const* d_in, const int* in_sizes, int n_in,
                              void* d_out, int out_size, void* d_ws, size_t ws_size,
                              hipStream_t stream) {
  const float* x      = (const float*)d_in[0];
  const float* freqs  = (const float*)d_in[1];
  // d_in[2] mask, d_in[3] input_pos: causal mask computed analytically
  const float* anw    = (const float*)d_in[4];
  const float* wqkv   = (const float*)d_in[5];
  const float* wo     = (const float*)d_in[6];
  const float* fnw    = (const float*)d_in[7];
  const float* w1     = (const float*)d_in[8];
  const float* w2     = (const float*)d_in[9];
  const float* w3     = (const float*)d_in[10];
  float* out          = (float*)d_out;

  char* ws = (char*)d_ws;
  _Float16* wqkv_h = (_Float16*)(ws + 0);           // 6144x2048
  _Float16* wo_h   = (_Float16*)(ws + 25165824);    // 2048x2048
  _Float16* w1_h   = (_Float16*)(ws + 33554432);    // 8192x2048
  _Float16* w3_h   = (_Float16*)(ws + 67108864);    // 8192x2048
  _Float16* w2_h   = (_Float16*)(ws + 100663296);   // 2048x8192
  _Float16* nx_h   = (_Float16*)(ws + 134217728);   // 4096x2048 (reused as nh)
  _Float16* qkv_h  = (_Float16*)(ws + 150994944);   // 4096x6144 (reused as attn out)
  _Float16* q_h    = (_Float16*)(ws + 201326592);   // (B,H,S,128)
  _Float16* k_h    = (_Float16*)(ws + 218103808);   // (B,H,S,128)
  _Float16* v_t    = (_Float16*)(ws + 234881024);   // (B,H,128,S)
  float*    h_f    = (float*)   (ws + 251658240);   // 4096x2048 f32
  _Float16* h1_h   = (_Float16*)(ws + 285212672);   // 4096x8192
  _Float16* h3_h   = (_Float16*)(ws + 352321536);   // 4096x8192
  _Float16* nh_h   = nx_h;
  _Float16* a_h    = qkv_h;

  // 1. weights f32 -> f16
  cvt_f32_f16<<<2048, 256, 0, stream>>>(wqkv, wqkv_h, 6144 * 2048);
  cvt_f32_f16<<<2048, 256, 0, stream>>>(wo,   wo_h,   2048 * 2048);
  cvt_f32_f16<<<2048, 256, 0, stream>>>(w1,   w1_h,   8192 * 2048);
  cvt_f32_f16<<<2048, 256, 0, stream>>>(w3,   w3_h,   8192 * 2048);
  cvt_f32_f16<<<2048, 256, 0, stream>>>(w2,   w2_h,   2048 * 8192);

  // 2. rmsnorm(x) -> nx
  rmsnorm_f16<<<4096, 256, 0, stream>>>(x, anw, nx_h, 2048);

  // 3. qkv = nx @ wqkv^T   (4096x2048)x(6144x2048)^T ; block tile 128x256
  gemm_wmma<<<dim3(24, 32), 256, 0, stream>>>(nx_h, wqkv_h, nullptr, nullptr, qkv_h, 6144, 2048);

  // 4. rope + v transpose
  rope_split<<<dim3(2048, 16, 2), 128, 0, stream>>>(qkv_h, freqs, q_h, k_h, v_t);

  // 5. flash attention -> a (4096x2048 f16)
  flash_attn<<<dim3(32, 16, 2), 128, 0, stream>>>(q_h, k_h, v_t, a_h);

  // 6. h = x + a @ wo^T
  gemm_wmma<<<dim3(8, 32), 256, 0, stream>>>(a_h, wo_h, x, h_f, nullptr, 2048, 2048);

  // 7. nh = rmsnorm(h)
  rmsnorm_f16<<<4096, 256, 0, stream>>>(h_f, fnw, nh_h, 2048);

  // 8/9. h1 = nh @ w1^T ; h3 = nh @ w3^T
  gemm_wmma<<<dim3(32, 32), 256, 0, stream>>>(nh_h, w1_h, nullptr, nullptr, h1_h, 8192, 2048);
  gemm_wmma<<<dim3(32, 32), 256, 0, stream>>>(nh_h, w3_h, nullptr, nullptr, h3_h, 8192, 2048);

  // 10. act = silu(h1) * h3  (in place into h1)
  silu_mul_f16<<<8192, 256, 0, stream>>>(h1_h, h3_h, h1_h, 4096 * 8192);

  // 11. out = h + act @ w2^T
  gemm_wmma<<<dim3(8, 32), 256, 0, stream>>>(h1_h, w2_h, h_f, out, nullptr, 2048, 8192);
}
